// UNet_19035295055997
// MI455X (gfx1250) — matrix operand order repaired and checked by reading, not compiled, not run on hardware
//
#include <hip/hip_runtime.h>
#include <hip/hip_bf16.h>

typedef __attribute__((ext_vector_type(16))) _Float16 v16h;
typedef __attribute__((ext_vector_type(8)))  float    v8f;
typedef __attribute__((ext_vector_type(4)))  float    v4f;

#define BATCH 4
#define CCH   256
#define HH    64
#define WW    64
#define LTOK  4096          // H*W
#define NTOK  16384         // B*L
#define C2CH  128
#define HIDN  1024

// ---------------------------------------------------------------------------
// CDNA5 async global->LDS copy (16 bytes), tracked by ASYNCcnt.
// LDS destination: low 32 bits of the generic shared-memory address are the
// LDS byte offset.
// ---------------------------------------------------------------------------
__device__ __forceinline__ void async_cp16(const _Float16* g, _Float16* l) {
    unsigned int loff = (unsigned int)(size_t)(void*)l;
    unsigned long long ga = (unsigned long long)(size_t)g;
    asm volatile("global_load_async_to_lds_b128 %0, %1, off"
                 :: "v"(loff), "v"(ga) : "memory");
}
__device__ __forceinline__ void wait_async0() {
    asm volatile("s_wait_asynccnt 0x0" ::: "memory");
}

// ---------------------------------------------------------------------------
// f32 -> f16 weight conversion
// ---------------------------------------------------------------------------
__global__ __launch_bounds__(256) void cvt_f16_kernel(const float* __restrict__ in,
                                                      _Float16* __restrict__ out, int n) {
    int i = blockIdx.x * 256 + threadIdx.x;
    if (i < n) out[i] = (_Float16)in[i];
}

// ---------------------------------------------------------------------------
// LayerNorm 1: x (B,C,H,W) f32 -> xt (B,L,C) f32 (residual base) + imgh f16
// ---------------------------------------------------------------------------
__global__ __launch_bounds__(256) void ln1_kernel(const float* __restrict__ x,
                                                  const float* __restrict__ w,
                                                  const float* __restrict__ bb,
                                                  float* __restrict__ xt,
                                                  _Float16* __restrict__ imgh) {
    int token = blockIdx.x * 256 + threadIdx.x;       // 0..16383
    int b = token >> 12;
    int l = token & (LTOK - 1);
    const float* xp = x + (size_t)b * CCH * LTOK + l;
    float s = 0.f, s2 = 0.f;
    for (int c = 0; c < CCH; ++c) {
        float v = xp[(size_t)c * LTOK];
        s += v; s2 += v * v;
    }
    float m  = s * (1.0f / CCH);
    float var = s2 * (1.0f / CCH) - m * m;
    float rs = rsqrtf(var + 1e-5f);
    float* xo = xt + (size_t)token * CCH;
    _Float16* ho = imgh + (size_t)token * CCH;
    for (int c = 0; c < CCH; ++c) {
        float v = xp[(size_t)c * LTOK];
        xo[c] = v;
        ho[c] = (_Float16)((v - m) * rs * w[c] + bb[c]);
    }
}

// ---------------------------------------------------------------------------
// LayerNorm 2: xin (NTOK,C) f32 token-major -> yh f16. One wave per token.
// ---------------------------------------------------------------------------
__global__ __launch_bounds__(256) void ln2_kernel(const float* __restrict__ xin,
                                                  const float* __restrict__ w,
                                                  const float* __restrict__ bb,
                                                  _Float16* __restrict__ yh) {
    int token = blockIdx.x * 8 + (threadIdx.x >> 5);
    int lane  = threadIdx.x & 31;
    const float* xp = xin + (size_t)token * CCH;
    float vals[8];
    float s = 0.f, s2 = 0.f;
#pragma unroll
    for (int i = 0; i < 8; ++i) {
        float v = xp[lane + 32 * i];
        vals[i] = v; s += v; s2 += v * v;
    }
#pragma unroll
    for (int m = 1; m < 32; m <<= 1) {
        s  += __shfl_xor(s,  m, 32);
        s2 += __shfl_xor(s2, m, 32);
    }
    float mean = s * (1.0f / CCH);
    float var  = s2 * (1.0f / CCH) - mean * mean;
    float rs   = rsqrtf(var + 1e-5f);
    _Float16* yo = yh + (size_t)token * CCH;
#pragma unroll
    for (int i = 0; i < 8; ++i) {
        int c = lane + 32 * i;
        yo[c] = (_Float16)((vals[i] - mean) * rs * w[c] + bb[c]);
    }
}

// ---------------------------------------------------------------------------
// WMMA GEMM:  out = act(A (M,K) @ W(N,K)^T + bias) [+ res]
// 128x128 block tile, BK=32, 8 waves -> each wave 32x64 (8 WMMAs / K-step).
// Double-buffered LDS fed by async global->LDS copies (ASYNCcnt).
// ---------------------------------------------------------------------------
template <bool HAS_BIAS, bool GELU_ACT, bool RES_ADD, bool OUT16>
__global__ __launch_bounds__(256) void gemm_wmma(const _Float16* __restrict__ A,
                                                 const _Float16* __restrict__ W,
                                                 const float* __restrict__ bias,
                                                 const float* __restrict__ res,
                                                 float* __restrict__ outF,
                                                 _Float16* __restrict__ outH,
                                                 int M, int N, int K) {
    __shared__ _Float16 As[2][128][40];
    __shared__ _Float16 Bs[2][128][40];
    const int tid  = threadIdx.x;
    const int lane = tid & 31;
    const int wave = tid >> 5;
    const int bm = blockIdx.x * 128;
    const int bn = blockIdx.y * 128;
    const int wm = (wave & 3) * 32;   // 4 waves along M, 32 rows each
    const int wn = (wave >> 2) * 64;  // 2 waves along N, 64 cols each
    const int hi = lane >> 4, lo = lane & 15;

    v8f acc[2][4];
#pragma unroll
    for (int mi = 0; mi < 2; ++mi)
#pragma unroll
        for (int nj = 0; nj < 4; ++nj) acc[mi][nj] = (v8f){};

    // tile staging: 128 rows x 32 halves = 512 16B chunks per matrix,
    // 256 threads -> 2 chunks each, issued as async LDS copies.
    auto stage = [&](int p, int k0) {
#pragma unroll
        for (int c = 0; c < 2; ++c) {
            int ch  = tid * 2 + c;
            int row = ch >> 2;
            int seg = (ch & 3) * 8;
            async_cp16(&A[(size_t)(bm + row) * K + k0 + seg], &As[p][row][seg]);
            async_cp16(&W[(size_t)(bn + row) * K + k0 + seg], &Bs[p][row][seg]);
        }
    };

    auto compute = [&](int p) {
        v16h af[2], bf[4];
#pragma unroll
        for (int mi = 0; mi < 2; ++mi) {
            // A fragment: lane group hi holds K slices {hi*8..+7},{16+hi*8..+7}
            *((v4f*)&af[mi])     = *(const v4f*)&As[p][wm + mi * 16 + lo][hi * 8];
            *((v4f*)&af[mi] + 1) = *(const v4f*)&As[p][wm + mi * 16 + lo][16 + hi * 8];
        }
#pragma unroll
        for (int nj = 0; nj < 4; ++nj) {
            // B fragment: lane group hi holds K slices {hi*8..+7},{16+hi*8..+7}
            *((v4f*)&bf[nj])     = *(const v4f*)&Bs[p][wn + nj * 16 + lo][hi * 8];
            *((v4f*)&bf[nj] + 1) = *(const v4f*)&Bs[p][wn + nj * 16 + lo][16 + hi * 8];
        }
#pragma unroll
        for (int mi = 0; mi < 2; ++mi)
#pragma unroll
            for (int nj = 0; nj < 4; ++nj)
                acc[mi][nj] = __builtin_amdgcn_wmma_f32_16x16x32_f16(
                    false, af[mi], false, bf[nj], (short)0, acc[mi][nj], false, false);
    };

    int p = 0;
    stage(0, 0);
    wait_async0();
    __syncthreads();
    for (int k0 = 32; k0 < K; k0 += 32) {
        stage(p ^ 1, k0);   // async copies in flight during compute
        compute(p);
        wait_async0();
        __syncthreads();    // waits dscnt too: all frag reads of buf p done
        p ^= 1;
    }
    compute(p);

#pragma unroll
    for (int mi = 0; mi < 2; ++mi) {
#pragma unroll
        for (int nj = 0; nj < 4; ++nj) {
#pragma unroll
            for (int r = 0; r < 8; ++r) {
                int row = bm + wm + mi * 16 + hi * 8 + r;
                int col = bn + wn + nj * 16 + lo;
                float v = acc[mi][nj][r];
                if (HAS_BIAS) v += bias[col];
                if (GELU_ACT) v = 0.5f * v * (1.0f + erff(v * 0.70710678118654752f));
                if (RES_ADD)  v += res[(size_t)row * N + col];
                if (OUT16) outH[(size_t)row * N + col] = (_Float16)v;
                else       outF[(size_t)row * N + col] = v;
            }
        }
    }
}

// ---------------------------------------------------------------------------
// token mapping: window token t -> global token index
// ---------------------------------------------------------------------------
template <int BR>
__device__ __forceinline__ int lglob(int b, int win, int t) {
    if (BR == 0) return b * LTOK + (t >> 3) * WW + win * 8 + (t & 7);
    else         return b * LTOK + win * 512 + t;
}

// ---------------------------------------------------------------------------
// LePE depthwise 3x3 conv per window (SAME padding inside window)
// ---------------------------------------------------------------------------
template <int BR>
__global__ __launch_bounds__(256) void lepe_kernel(const _Float16* __restrict__ qkv,
                                                   const float* __restrict__ gw,
                                                   const float* __restrict__ gb,
                                                   float* __restrict__ lepe) {
    size_t idx = (size_t)blockIdx.x * 256 + threadIdx.x;  // < 16384*128
    int ch = (int)(idx & 127);
    int tk = (int)(idx >> 7);                             // 0..16383
    int b  = tk >> 12;
    int lw = tk & (LTOK - 1);
    int row = lw >> 6;       // 0..63
    int colg = lw & 63;      // 0..63
    int win, hs, ws, Hsp, Wsp;
    if (BR == 0) { win = colg >> 3; ws = colg & 7; hs = row;    Hsp = 64; Wsp = 8; }
    else         { win = row >> 3;  hs = row & 7;  ws = colg;   Hsp = 8;  Wsp = 64; }
    float acc = gb[ch];
#pragma unroll
    for (int dy = -1; dy <= 1; ++dy) {
#pragma unroll
        for (int dx = -1; dx <= 1; ++dx) {
            int y = hs + dy, x = ws + dx;
            if (y >= 0 && y < Hsp && x >= 0 && x < Wsp) {
                int l2;
                if (BR == 0) l2 = b * LTOK + y * WW + win * 8 + x;
                else         l2 = b * LTOK + (win * 8 + y) * WW + x;
                float vv = (float)qkv[(size_t)l2 * 512 + 256 + BR * C2CH + ch];
                acc += vv * gw[ch * 9 + (dy + 1) * 3 + (dx + 1)];
            }
        }
    }
    lepe[(size_t)tk * CCH + BR * C2CH + ch] = acc;
}

// ---------------------------------------------------------------------------
// Windowed attention (flash style), one block per (b, window, head).
// seq = 512, hd = 32. V staged transposed in LDS; Q/K fragments from global.
// ---------------------------------------------------------------------------
template <int BR>
__global__ __launch_bounds__(256) void attn_kernel(const _Float16* __restrict__ qkv,
                                                   const float* __restrict__ lepe,
                                                   _Float16* __restrict__ attO) {
    __shared__ _Float16 Vt[32][520];        // V^T: [dim][token]
    __shared__ _Float16 Pbuf[8][16][40];    // per-wave P round-trip

    const int blk  = blockIdx.x;            // 128 blocks
    const int head = blk & 3;
    const int win  = (blk >> 2) & 7;
    const int b    = blk >> 5;
    const int col0 = BR * C2CH + head * 32; // qk channel base

    const int tid  = threadIdx.x;
    const int lane = tid & 31;
    const int wave = tid >> 5;
    const int hi   = lane >> 4, lo = lane & 15;

    for (int t = tid; t < 512; t += 256) {
        int l = lglob<BR>(b, win, t);
        const _Float16* vp = &qkv[(size_t)l * 512 + 256 + col0];
#pragma unroll
        for (int d = 0; d < 32; ++d) Vt[d][t] = vp[d];
    }
    __syncthreads();

    const float scale = 0.17677669529663687f;  // 1/sqrt(32)

    for (int qi = 0; qi < 4; ++qi) {
        const int qt = (qi * 8 + wave) * 16;   // q-tile base row in window
        v16h qf;
        {
            int ql = lglob<BR>(b, win, qt + lo);
            const _Float16* qp = &qkv[(size_t)ql * 512 + col0];
            *((v4f*)&qf)     = *(const v4f*)(qp + hi * 8);
            *((v4f*)&qf + 1) = *(const v4f*)(qp + 16 + hi * 8);
        }

        float Mr[8], Lr[8];
        v8f o0 = {}, o1 = {};
#pragma unroll
        for (int r = 0; r < 8; ++r) { Mr[r] = -1e30f; Lr[r] = 0.f; }

        for (int kc = 0; kc < 16; ++kc) {
            const int kbase = kc * 32;
            v16h kf0, kf1;
            {
                int k0l = lglob<BR>(b, win, kbase + lo);
                int k1l = lglob<BR>(b, win, kbase + 16 + lo);
                const _Float16* kp0 = &qkv[(size_t)k0l * 512 + col0];
                const _Float16* kp1 = &qkv[(size_t)k1l * 512 + col0];
                *((v4f*)&kf0)     = *(const v4f*)(kp0 + hi * 8);
                *((v4f*)&kf0 + 1) = *(const v4f*)(kp0 + 16 + hi * 8);
                *((v4f*)&kf1)     = *(const v4f*)(kp1 + hi * 8);
                *((v4f*)&kf1 + 1) = *(const v4f*)(kp1 + 16 + hi * 8);
            }
            v8f s0 = {}, s1 = {};
            s0 = __builtin_amdgcn_wmma_f32_16x16x32_f16(false, qf, false, kf0,
                                                        (short)0, s0, false, false);
            s1 = __builtin_amdgcn_wmma_f32_16x16x32_f16(false, qf, false, kf1,
                                                        (short)0, s1, false, false);

#pragma unroll
            for (int r = 0; r < 8; ++r) {
                float a0 = s0[r] * scale, a1 = s1[r] * scale;
                float mx = fmaxf(a0, a1);
#pragma unroll
                for (int m = 1; m < 16; m <<= 1) mx = fmaxf(mx, __shfl_xor(mx, m, 32));
                float nm   = fmaxf(Mr[r], mx);
                float corr = __expf(Mr[r] - nm);
                float e0 = __expf(a0 - nm);
                float e1 = __expf(a1 - nm);
                float rs = e0 + e1;
#pragma unroll
                for (int m = 1; m < 16; m <<= 1) rs += __shfl_xor(rs, m, 32);
                Lr[r] = Lr[r] * corr + rs;
                Mr[r] = nm;
                o0[r] *= corr;
                o1[r] *= corr;
                Pbuf[wave][r + hi * 8][lo]      = (_Float16)e0;
                Pbuf[wave][r + hi * 8][16 + lo] = (_Float16)e1;
            }
            __syncthreads();   // uniform loops across all 8 waves

            v16h pf;
            *((v4f*)&pf)     = *(const v4f*)&Pbuf[wave][lo][hi * 8];
            *((v4f*)&pf + 1) = *(const v4f*)&Pbuf[wave][lo][16 + hi * 8];
            v16h vf0, vf1;
            *((v4f*)&vf0)     = *(const v4f*)&Vt[lo][kbase + hi * 8];
            *((v4f*)&vf0 + 1) = *(const v4f*)&Vt[lo][kbase + 16 + hi * 8];
            *((v4f*)&vf1)     = *(const v4f*)&Vt[16 + lo][kbase + hi * 8];
            *((v4f*)&vf1 + 1) = *(const v4f*)&Vt[16 + lo][kbase + 16 + hi * 8];

            o0 = __builtin_amdgcn_wmma_f32_16x16x32_f16(false, pf, false, vf0,
                                                        (short)0, o0, false, false);
            o1 = __builtin_amdgcn_wmma_f32_16x16x32_f16(false, pf, false, vf1,
                                                        (short)0, o1, false, false);
        }

#pragma unroll
        for (int r = 0; r < 8; ++r) {
            int t = qt + r + hi * 8;
            int l = lglob<BR>(b, win, t);
            float inv = 1.0f / Lr[r];
            size_t base = (size_t)l * CCH + col0;
            float v0 = o0[r] * inv + lepe[base + lo];
            float v1 = o1[r] * inv + lepe[base + 16 + lo];
            attO[base + lo]      = (_Float16)v0;
            attO[base + 16 + lo] = (_Float16)v1;
        }
    }
}

// ---------------------------------------------------------------------------
// launch
// ---------------------------------------------------------------------------
extern "C" void kernel_launch(void* const* d_in, const int* in_sizes, int n_in,
                              void* d_out, int out_size, void* d_ws, size_t ws_size,
                              hipStream_t stream) {
    const float* x       = (const float*)d_in[0];
    const float* n1w     = (const float*)d_in[1];
    const float* n1b     = (const float*)d_in[2];
    const float* qkv_w   = (const float*)d_in[3];
    const float* gw0     = (const float*)d_in[4];
    const float* gb0     = (const float*)d_in[5];
    const float* gw1     = (const float*)d_in[6];
    const float* gb1     = (const float*)d_in[7];
    const float* proj_w  = (const float*)d_in[8];
    const float* proj_b  = (const float*)d_in[9];
    const float* n2w     = (const float*)d_in[10];
    const float* n2b     = (const float*)d_in[11];
    const float* fc1_w   = (const float*)d_in[12];
    const float* fc1_b   = (const float*)d_in[13];
    const float* fc2_w   = (const float*)d_in[14];
    const float* fc2_b   = (const float*)d_in[15];
    float* out = (float*)d_out;

    size_t off = 0;
    auto alloc = [&](size_t bytes) {
        void* p = (char*)d_ws + off;
        off += (bytes + 255) & ~(size_t)255;
        return p;
    };
    _Float16* wqkvh = (_Float16*)alloc((size_t)512 * 256 * 2);
    _Float16* wprjh = (_Float16*)alloc((size_t)256 * 256 * 2);
    _Float16* wfc1h = (_Float16*)alloc((size_t)1024 * 256 * 2);
    _Float16* wfc2h = (_Float16*)alloc((size_t)256 * 1024 * 2);
    float*    xt    = (float*)   alloc((size_t)NTOK * CCH * 4);
    _Float16* imgh  = (_Float16*)alloc((size_t)NTOK * CCH * 2);
    _Float16* qkvh  = (_Float16*)alloc((size_t)NTOK * 512 * 2);
    float*    lepe  = (float*)   alloc((size_t)NTOK * CCH * 4);
    _Float16* atth  = (_Float16*)alloc((size_t)NTOK * CCH * 2);
    float*    xnew  = (float*)   alloc((size_t)NTOK * CCH * 4);
    _Float16* yh    = (_Float16*)alloc((size_t)NTOK * CCH * 2);
    _Float16* hidh  = (_Float16*)alloc((size_t)NTOK * HIDN * 2);

    // 1. convert weights to f16
    cvt_f16_kernel<<<512 * 256 / 256, 256, 0, stream>>>(qkv_w, wqkvh, 512 * 256);
    cvt_f16_kernel<<<256 * 256 / 256, 256, 0, stream>>>(proj_w, wprjh, 256 * 256);
    cvt_f16_kernel<<<1024 * 256 / 256, 256, 0, stream>>>(fc1_w, wfc1h, 1024 * 256);
    cvt_f16_kernel<<<256 * 1024 / 256, 256, 0, stream>>>(fc2_w, wfc2h, 256 * 1024);

    // 2. LN1 (+ transpose residual base)
    ln1_kernel<<<NTOK / 256, 256, 0, stream>>>(x, n1w, n1b, xt, imgh);

    // 3. qkv GEMM -> f16
    gemm_wmma<false, false, false, true><<<dim3(NTOK / 128, 512 / 128), 256, 0, stream>>>(
        imgh, wqkvh, nullptr, nullptr, nullptr, qkvh, NTOK, 512, 256);

    // 4. LePE depthwise conv (both branches)
    lepe_kernel<0><<<(NTOK * 128) / 256, 256, 0, stream>>>(qkvh, gw0, gb0, lepe);
    lepe_kernel<1><<<(NTOK * 128) / 256, 256, 0, stream>>>(qkvh, gw1, gb1, lepe);

    // 5. windowed attention (both branches) -> atth f16
    attn_kernel<0><<<128, 256, 0, stream>>>(qkvh, lepe, atth);
    attn_kernel<1><<<128, 256, 0, stream>>>(qkvh, lepe, atth);

    // 6. proj GEMM + bias + residual -> xnew f32
    gemm_wmma<true, false, true, false><<<dim3(NTOK / 128, 256 / 128), 256, 0, stream>>>(
        atth, wprjh, proj_b, xt, xnew, nullptr, NTOK, 256, 256);

    // 7. LN2
    ln2_kernel<<<NTOK / 8, 256, 0, stream>>>(xnew, n2w, n2b, yh);

    // 8. fc1 GEMM + bias + GELU -> hidh f16
    gemm_wmma<true, true, false, true><<<dim3(NTOK / 128, 1024 / 128), 256, 0, stream>>>(
        yh, wfc1h, fc1_b, nullptr, nullptr, hidh, NTOK, 1024, 256);

    // 9. fc2 GEMM + bias + residual -> final output f32
    gemm_wmma<true, false, true, false><<<dim3(NTOK / 128, 256 / 128), 256, 0, stream>>>(
        hidh, wfc2h, fc2_b, xnew, out, nullptr, NTOK, 256, HIDN);
}
